// NeighborDiscriminator_49898930045648
// MI455X (gfx1250) — matrix operand order repaired and checked by reading, not compiled, not run on hardware
//
#include <hip/hip_runtime.h>

#define KCONST 750.0f
#define TOPK 16

typedef __attribute__((ext_vector_type(16))) __bf16 v16bf;
typedef __attribute__((ext_vector_type(8)))  float  v8f;
typedef __attribute__((ext_vector_type(4)))  float  f32x4;

union FragU { v16bf v; __bf16 e[16]; };

// ---------------------------------------------------------------------------
// Kernel 0: max(w) reduction
// ---------------------------------------------------------------------------
__global__ __launch_bounds__(256) void wmax_kernel(const float* __restrict__ w,
                                                   int n, float* __restrict__ wmax_out) {
    __shared__ float red[256];
    int tid = threadIdx.x;
    float m = -3.4e38f;
    for (int j = tid; j < n; j += 256) m = fmaxf(m, w[j]);
    red[tid] = m;
    __syncthreads();
    for (int s = 128; s > 0; s >>= 1) {
        if (tid < s) red[tid] = fmaxf(red[tid], red[tid + s]);
        __syncthreads();
    }
    if (tid == 0) *wmax_out = red[0];
}

// ---------------------------------------------------------------------------
// Kernel 1: score[i,j] = (|x_j|^2 + w'_j^2) - 2 * (q_i . x_j)
//   bf16 WMMA GEMM (f32 accumulate). X is streamed once with NT loads so the
//   score matrix (written RT) stays resident in the 192MB L2 for the top-k
//   pass, and X_tilde (re-read by every block) stays cached.
//   Block: 256 threads = 8 waves as 4(M) x 2(N). Tile: 256(M) x 64(N), K step 32.
// ---------------------------------------------------------------------------
__global__ __launch_bounds__(256) void score_kernel(
    const float* __restrict__ Xt, const float* __restrict__ X,
    const float* __restrict__ w,  const float* __restrict__ wmax_p,
    float* __restrict__ score, int m, int n, int d, int Npad) {

    const int tid  = threadIdx.x;
    const int lane = tid & 31;
    const int wv   = tid >> 5;
    const int wm   = wv & 3;            // M group (0..3)
    const int wn   = wv >> 2;           // N group (0..1)
    const int l15  = lane & 15;
    const int hi   = (lane >> 4) & 1;   // half-wave select

    const int nb = blockIdx.x * 64;     // database-column base
    const int mb = blockIdx.y * 256;    // query-row base

    __shared__ float n2s[64];

    const v8f vzero = {0.f, 0.f, 0.f, 0.f, 0.f, 0.f, 0.f, 0.f};
    v8f acc[4][2];
#pragma unroll
    for (int a = 0; a < 4; ++a)
#pragma unroll
        for (int b = 0; b < 2; ++b) acc[a][b] = vzero;

    // A fragment row pointers (lane -> query row), clamped
    const float* rA[4];
#pragma unroll
    for (int a = 0; a < 4; ++a) {
        int r = mb + wm * 64 + a * 16 + l15;
        if (r > m - 1) r = m - 1;
        rA[a] = Xt + (size_t)r * d;
    }
    // B fragment row pointers (lane -> database row), clamped
    const float* rB[2];
#pragma unroll
    for (int b = 0; b < 2; ++b) {
        int j = nb + wn * 32 + b * 16 + l15;
        if (j > n - 1) j = n - 1;
        rB[b] = X + (size_t)j * d;
    }

    const int kA = hi ? 16 : 0;   // A: one 16-elem chunk at kA
    const int kB = hi ? 16 : 0;   // B: one 16-elem chunk at kB

    float sq[2] = {0.f, 0.f};     // fused |x_j|^2 partials (f32, pre-conversion)

#pragma unroll 2
    for (int k0 = 0; k0 < d; k0 += 32) {
        FragU fb[2];
#pragma unroll
        for (int b = 0; b < 2; ++b) {
            const float* p = rB[b] + k0 + kB;
            float tb[16];
            // X is touched exactly once: non-temporal so it does not evict
            // X_tilde or the score matrix from L2.
            *(f32x4*)(tb + 0)  = __builtin_nontemporal_load((const f32x4*)(p + 0));
            *(f32x4*)(tb + 4)  = __builtin_nontemporal_load((const f32x4*)(p + 4));
            *(f32x4*)(tb + 8)  = __builtin_nontemporal_load((const f32x4*)(p + 8));
            *(f32x4*)(tb + 12) = __builtin_nontemporal_load((const f32x4*)(p + 12));
            if (wm == 0) {        // wave-uniform: only norm-owning waves pay the FMAs
#pragma unroll
                for (int i = 0; i < 16; ++i) sq[b] += tb[i] * tb[i];
            }
#pragma unroll
            for (int i = 0; i < 16; ++i) fb[b].e[i] = (__bf16)tb[i];
        }
#pragma unroll
        for (int a = 0; a < 4; ++a) {
            const float* p = rA[a] + k0 + kA;
            float ta[16];
            *(f32x4*)(ta + 0)  = *(const f32x4*)(p + 0);
            *(f32x4*)(ta + 4)  = *(const f32x4*)(p + 4);
            *(f32x4*)(ta + 8)  = *(const f32x4*)(p + 8);
            *(f32x4*)(ta + 12) = *(const f32x4*)(p + 12);
            FragU fa;
#pragma unroll
            for (int i = 0; i < 16; ++i) fa.e[i] = (__bf16)ta[i];
#pragma unroll
            for (int b = 0; b < 2; ++b)
                acc[a][b] = __builtin_amdgcn_wmma_f32_16x16x32_bf16(
                    false, fa.v, false, fb[b].v, (short)0, acc[a][b], false, false);
        }
    }

    // Fused row norms: combine half-wave partials, add (w'-term)^2, share via LDS
    const float wmax = *wmax_p;
#pragma unroll
    for (int b = 0; b < 2; ++b) {
        float tot = sq[b] + __shfl_xor(sq[b], 16, 32);
        if (wm == 0 && lane < 16) {
            int j = nb + wn * 32 + b * 16 + lane;
            float n2 = tot;
            if (j < n) {
                float wp = (w[j] - wmax) * (1.0f / KCONST);
                n2 += wp * wp;
            }
            n2s[wn * 32 + b * 16 + lane] = n2;
        }
    }
    __syncthreads();

    // Epilogue: C layout -> lanes 0-15: N=lane, M=v ; lanes 16-31: N=lane-16, M=v+8
#pragma unroll
    for (int a = 0; a < 4; ++a) {
        const int ibase = mb + wm * 64 + a * 16 + hi * 8;
#pragma unroll
        for (int b = 0; b < 2; ++b) {
            const int jl = wn * 32 + b * 16 + l15;
            const int j  = nb + jl;
            const float n2 = n2s[jl];
            const bool jvalid = (j < n);
#pragma unroll
            for (int v = 0; v < 8; ++v) {
                int i = ibase + v;
                if (i < m) {
                    float s = jvalid ? (n2 - 2.0f * acc[a][b][v]) : 3.0e38f;
                    score[(size_t)i * Npad + j] = s;
                }
            }
        }
    }
}

// ---------------------------------------------------------------------------
// Kernel 2: per-query top-16 (smallest score, stable ties by index)
// ---------------------------------------------------------------------------
__global__ __launch_bounds__(256) void topk_kernel(const float* __restrict__ score,
                                                   int n, int Npad,
                                                   int* __restrict__ topk) {
    const int i   = blockIdx.x;
    const int tid = threadIdx.x;
    const float* row = score + (size_t)i * Npad;

    float lv[TOPK];
    int   li[TOPK];
#pragma unroll
    for (int t = 0; t < TOPK; ++t) { lv[t] = 3.4e38f; li[t] = 0x7fffffff; }

    for (int j = tid; j < Npad; j += 256) {
        float v = row[j];
        if (v < lv[TOPK - 1] || (v == lv[TOPK - 1] && j < li[TOPK - 1])) {
            int p = TOPK - 1;
            while (p > 0 && (v < lv[p - 1] || (v == lv[p - 1] && j < li[p - 1]))) {
                lv[p] = lv[p - 1]; li[p] = li[p - 1]; --p;
            }
            lv[p] = v; li[p] = j;
        }
    }

    __shared__ float sv[256 * TOPK];
    __shared__ int   si[256 * TOPK];
#pragma unroll
    for (int t = 0; t < TOPK; ++t) { sv[tid * TOPK + t] = lv[t]; si[tid * TOPK + t] = li[t]; }
    __syncthreads();

    for (int s = 128; s > 0; s >>= 1) {
        if (tid < s) {
            const float* ov = &sv[(tid + s) * TOPK];
            const int*   oi = &si[(tid + s) * TOPK];
            float rv[TOPK]; int ri[TOPK];
            int pa = 0, pb = 0;
#pragma unroll
            for (int o = 0; o < TOPK; ++o) {
                bool ta;
                if (pb >= TOPK) ta = true;
                else if (pa >= TOPK) ta = false;
                else ta = (lv[pa] < ov[pb]) || (lv[pa] == ov[pb] && li[pa] <= oi[pb]);
                if (ta) { rv[o] = lv[pa]; ri[o] = li[pa]; ++pa; }
                else    { rv[o] = ov[pb]; ri[o] = oi[pb]; ++pb; }
            }
#pragma unroll
            for (int o = 0; o < TOPK; ++o) {
                lv[o] = rv[o]; li[o] = ri[o];
                sv[tid * TOPK + o] = rv[o]; si[tid * TOPK + o] = ri[o];
            }
        }
        __syncthreads();
    }

    if (tid == 0) {
#pragma unroll
        for (int t = 0; t < TOPK; ++t) topk[i * TOPK + t] = li[t];
    }
}

// ---------------------------------------------------------------------------
// Kernel 3: exact L1 refinement, max activation (first-argmax tie-break)
// ---------------------------------------------------------------------------
__global__ __launch_bounds__(256) void refine_kernel(
    const float* __restrict__ Xt, const float* __restrict__ X,
    const float* __restrict__ w,  const int* __restrict__ topk,
    float* __restrict__ out, int d) {

    const int i   = blockIdx.x;
    const int tid = threadIdx.x;
    __shared__ float red[256];
    __shared__ float acts[TOPK];
    const float* q = Xt + (size_t)i * d;

    for (int t = 0; t < TOPK; ++t) {
        int j = topk[i * TOPK + t];
        const float* x = X + (size_t)j * d;
        float p = 0.f;
        for (int dd = tid; dd < d; dd += 256) p += fabsf(x[dd] - q[dd]);
        red[tid] = p;
        __syncthreads();
        for (int s = 128; s > 0; s >>= 1) {
            if (tid < s) red[tid] += red[tid + s];
            __syncthreads();
        }
        if (tid == 0) acts[t] = w[j] - KCONST * red[0];
        __syncthreads();
    }
    if (tid == 0) {
        float best = acts[0];
#pragma unroll
        for (int t = 1; t < TOPK; ++t)
            if (acts[t] > best) best = acts[t];
        out[i] = best;
    }
}

// ---------------------------------------------------------------------------
extern "C" void kernel_launch(void* const* d_in, const int* in_sizes, int n_in,
                              void* d_out, int out_size, void* d_ws, size_t ws_size,
                              hipStream_t stream) {
    const float* Xt = (const float*)d_in[0];
    const float* X  = (const float*)d_in[1];
    const float* w  = (const float*)d_in[2];
    float* out = (float*)d_out;

    const int m = out_size;                 // 256 queries
    const int d = in_sizes[0] / m;          // 3072
    const int n = in_sizes[2];              // 50000 database rows
    const int Npad = ((n + 63) / 64) * 64;  // padded column count

    float* wmaxp = (float*)d_ws;
    float* score = (float*)((char*)d_ws + 256);
    int*   topk  = (int*)((char*)d_ws + 256 + (size_t)m * Npad * sizeof(float));

    wmax_kernel<<<1, 256, 0, stream>>>(w, n, wmaxp);

    dim3 grid(Npad / 64, (m + 255) / 256);
    score_kernel<<<grid, 256, 0, stream>>>(Xt, X, w, wmaxp, score, m, n, d, Npad);

    topk_kernel<<<m, 256, 0, stream>>>(score, n, Npad, topk);

    refine_kernel<<<m, 256, 0, stream>>>(Xt, X, w, topk, out, d);
}
